// Block_83391085019939
// MI455X (gfx1250) — matrix operand order repaired and checked by reading, not compiled, not run on hardware
//
#include <hip/hip_runtime.h>
#include <cstdint>
#include <cstddef>

// ---------------------------------------------------------------------------
// CDNA5 (gfx1250) WMMA types & helpers
// ---------------------------------------------------------------------------
typedef __attribute__((ext_vector_type(16))) __bf16 v16bf;
typedef __attribute__((ext_vector_type(8)))  float  v8f;

union FragA { v16bf v; uint16_t h[16]; uint32_t u[8]; uint4 q[2]; };
union FragC { v8f v; float f[8]; };

__device__ __forceinline__ uint16_t f2bf(float f) {
  uint32_t u = __float_as_uint(f);
  return (uint16_t)((u + 0x7FFFu + ((u >> 16) & 1u)) >> 16);
}
__device__ __forceinline__ float geluf(float x) {
  return 0.5f * x * (1.f + erff(x * 0.70710678118654752f));
}
__device__ __forceinline__ float siluf(float x) {
  return x / (1.f + __expf(-1.702f * x));
}
__device__ __forceinline__ unsigned f32ord(float f) {
  unsigned u = __float_as_uint(f);
  return (u & 0x80000000u) ? ~u : (u | 0x80000000u);
}
__device__ __forceinline__ float ord2f(unsigned o) {
  unsigned u = (o & 0x80000000u) ? (o & 0x7FFFFFFFu) : ~o;
  return __uint_as_float(u);
}
__device__ __forceinline__ v8f wmma_bf16(const FragA& a, const FragA& b, v8f c) {
  return __builtin_amdgcn_wmma_f32_16x16x32_bf16(false, a.v, false, b.v,
                                                 (short)0, c, false, false);
}

// ---------------------------------------------------------------------------
// Weight convert + transpose:  W[K][N] (f32)  ->  Wt[N][K] (bf16)
// (B-fragment of v_wmma wants, per lane n, contiguous K runs.)
// ---------------------------------------------------------------------------
__global__ void wtrans_kernel(const float* __restrict__ W, uint16_t* __restrict__ Wt,
                              int K, int Nn) {
  int idx = blockIdx.x * blockDim.x + threadIdx.x;
  if (idx >= K * Nn) return;
  int k = idx / Nn, n = idx - k * Nn;
  Wt[(size_t)n * K + k] = f2bf(W[idx]);
}

// ---------------------------------------------------------------------------
// LayerNorm over rows of length 384 -> bf16 output (GEMM/attention input)
// One wave per row.
// ---------------------------------------------------------------------------
__global__ void __launch_bounds__(256) ln_kernel(
    const float* __restrict__ in, const float* __restrict__ gam,
    const float* __restrict__ bet, uint16_t* __restrict__ outB, int rows) {
  const int Cc = 384;
  const int lane = threadIdx.x & 31;
  const int row = blockIdx.x * 8 + (threadIdx.x >> 5);
  if (row >= rows) return;
  const float* p = in + (size_t)row * Cc;
  float s = 0.f, ss = 0.f;
#pragma unroll
  for (int j = 0; j < 12; ++j) { float t = p[lane + 32 * j]; s += t; ss += t * t; }
#pragma unroll
  for (int o = 1; o < 32; o <<= 1) { s += __shfl_xor(s, o, 32); ss += __shfl_xor(ss, o, 32); }
  const float mean = s * (1.f / 384.f);
  const float var  = ss * (1.f / 384.f) - mean * mean;
  const float rstd = rsqrtf(var + 1e-5f);
  uint16_t* op = outB + (size_t)row * Cc;
#pragma unroll
  for (int j = 0; j < 12; ++j) {
    int c = lane + 32 * j;
    op[c] = f2bf((p[c] - mean) * rstd * gam[c] + bet[c]);
  }
}

// ---------------------------------------------------------------------------
// Generic GEMM:  out[M,N] = act(A[M,K](bf16) x Bt[N,K](bf16) + bias) (+resid)
// One wave computes a 32x32 macro-tile: 2 A-frags x 2 B-frags -> 4 WMMAs per
// K-step (2x arithmetic intensity vs one tile/wave, 4 independent WMMAs for
// the scheduler to overlap with the b128 load clauses).
// act: 0=none  1=exact-GELU  2=silu(1.702x)
// ---------------------------------------------------------------------------
__global__ void __launch_bounds__(128) gemm_wmma_kernel(
    const uint16_t* __restrict__ A, const uint16_t* __restrict__ Bt,
    const float* __restrict__ bias, const float* __restrict__ resid,
    float* __restrict__ outF, uint16_t* __restrict__ outB,
    int M, int Nn, int K, int act) {
  const int lane = threadIdx.x & 31;
  const int wv   = threadIdx.x >> 5;
  const int nt = blockIdx.x;                 // 32-column macro tile
  const int mt = blockIdx.y * 4 + wv;        // 32-row macro tile
  const int mTiles = (M + 31) >> 5;
  if (mt >= mTiles) return;
  const int half = lane >> 4, ln = lane & 15;

  int ar0 = mt * 32 + ln;      if (ar0 > M - 1) ar0 = M - 1;
  int ar1 = mt * 32 + 16 + ln; if (ar1 > M - 1) ar1 = M - 1;
  const uint16_t* Ar0 = A  + (size_t)ar0 * K;
  const uint16_t* Ar1 = A  + (size_t)ar1 * K;
  const uint16_t* Bc0 = Bt + (size_t)(nt * 32 + ln) * K;
  const uint16_t* Bc1 = Bt + (size_t)(nt * 32 + 16 + ln) * K;

  FragC a00, a01, a10, a11;
#pragma unroll
  for (int r = 0; r < 8; ++r) { a00.f[r] = 0.f; a01.f[r] = 0.f; a10.f[r] = 0.f; a11.f[r] = 0.f; }

  const int nK = K >> 5;
  for (int kc = 0; kc < nK; ++kc) {
    const int kb = kc << 5;
    __builtin_prefetch(Ar0 + kb + 256, 0, 1);
    __builtin_prefetch(Ar1 + kb + 256, 0, 1);
    __builtin_prefetch(Bc0 + kb + 256, 0, 1);
    __builtin_prefetch(Bc1 + kb + 256, 0, 1);
    FragA af0, af1, bf0, bf1;
    af0.q[0] = *(const uint4*)(Ar0 + kb + half * 8);
    af0.q[1] = *(const uint4*)(Ar0 + kb + 16 + half * 8);
    af1.q[0] = *(const uint4*)(Ar1 + kb + half * 8);
    af1.q[1] = *(const uint4*)(Ar1 + kb + 16 + half * 8);
    bf0.q[0] = *(const uint4*)(Bc0 + kb + half * 8);
    bf0.q[1] = *(const uint4*)(Bc0 + kb + 16 + half * 8);
    bf1.q[0] = *(const uint4*)(Bc1 + kb + half * 8);
    bf1.q[1] = *(const uint4*)(Bc1 + kb + 16 + half * 8);
    a00.v = wmma_bf16(af0, bf0, a00.v);
    a01.v = wmma_bf16(af0, bf1, a01.v);
    a10.v = wmma_bf16(af1, bf0, a10.v);
    a11.v = wmma_bf16(af1, bf1, a11.v);
  }

#pragma unroll
  for (int sm = 0; sm < 2; ++sm) {
#pragma unroll
    for (int sn = 0; sn < 2; ++sn) {
      const FragC& acc = sm == 0 ? (sn == 0 ? a00 : a01) : (sn == 0 ? a10 : a11);
      const int ncol = nt * 32 + sn * 16 + ln;
      const float bv = bias ? bias[ncol] : 0.f;
#pragma unroll
      for (int r = 0; r < 8; ++r) {
        const int row = mt * 32 + sm * 16 + r + 8 * half;
        if (row < M) {
          float vv = acc.f[r] + bv;
          if (act == 1)      vv = geluf(vv);
          else if (act == 2) vv = siluf(vv);
          if (resid) vv += resid[(size_t)row * Nn + ncol];
          if (outF) outF[(size_t)row * Nn + ncol] = vv;
          if (outB) outB[(size_t)row * Nn + ncol] = f2bf(vv);
        }
      }
    }
  }
}

// ---------------------------------------------------------------------------
// Main multi-head flash attention (B=4, G1=2049, NH=6, hd=64).
// qkv: bf16 [B*G1, 1152] with col = s*384 + h*64 + d.  One wave = (b,h,qtile).
// ---------------------------------------------------------------------------
__global__ void __launch_bounds__(128) attn_main_kernel(
    const uint16_t* __restrict__ qkv, uint16_t* __restrict__ obf) {
  const int G1 = 2049, C3 = 1152, NQT = 129;
  const int lane = threadIdx.x & 31;
  const int wv = threadIdx.x >> 5;
  const int wid = blockIdx.x * 4 + wv;
  if (wid >= 4 * 6 * NQT) return;
  const int qt = wid % NQT;
  const int h  = (wid / NQT) % 6;
  const int b  = wid / (NQT * 6);
  const int half = lane >> 4, ln = lane & 15;
  __shared__ __align__(16) uint16_t sP[4][512];
  uint16_t* myP = sP[wv];

  int q0 = qt * 16 + ln; if (q0 >= G1) q0 = G1 - 1;
  const uint16_t* qrow = qkv + (size_t)(b * G1 + q0) * C3 + h * 64;
  FragA qa[2];
#pragma unroll
  for (int dc = 0; dc < 2; ++dc) {
    qa[dc].q[0] = *(const uint4*)(qrow + dc * 32 + half * 8);
    qa[dc].q[1] = *(const uint4*)(qrow + dc * 32 + 16 + half * 8);
  }
  const uint16_t* kb_ = qkv + (size_t)(b * G1) * C3 + 384 + h * 64;
  const uint16_t* vb_ = qkv + (size_t)(b * G1) * C3 + 768 + h * 64;
  float mst[8], lst[8];
  FragC acc[4];
#pragma unroll
  for (int r = 0; r < 8; ++r) { mst[r] = -3.0e38f; lst[r] = 0.f; }
#pragma unroll
  for (int d = 0; d < 4; ++d)
#pragma unroll
    for (int r = 0; r < 8; ++r) acc[d].f[r] = 0.f;

  for (int kc = 0; kc < 65; ++kc) {
    FragC S[2];
#pragma unroll
    for (int t = 0; t < 2; ++t) {
      const int tok = kc * 32 + t * 16 + ln;
      const int tc = tok < G1 ? tok : G1 - 1;
      const uint16_t* kp = kb_ + (size_t)tc * C3;
      FragC st;
#pragma unroll
      for (int r = 0; r < 8; ++r) st.f[r] = 0.f;
#pragma unroll
      for (int dc = 0; dc < 2; ++dc) {
        FragA kf;
        kf.q[0] = *(const uint4*)(kp + dc * 32 + half * 8);
        kf.q[1] = *(const uint4*)(kp + dc * 32 + 16 + half * 8);
        st.v = wmma_bf16(qa[dc], kf, st.v);
      }
      const bool valid = tok < G1;
#pragma unroll
      for (int r = 0; r < 8; ++r) st.f[r] = valid ? st.f[r] * 0.125f : -3.0e38f;
      S[t] = st;
    }
    float rm[8], al[8], rs[8];
#pragma unroll
    for (int r = 0; r < 8; ++r) rm[r] = fmaxf(S[0].f[r], S[1].f[r]);
#pragma unroll
    for (int o = 1; o < 16; o <<= 1)
#pragma unroll
      for (int r = 0; r < 8; ++r) rm[r] = fmaxf(rm[r], __shfl_xor(rm[r], o, 32));
#pragma unroll
    for (int r = 0; r < 8; ++r) {
      const float mn = fmaxf(mst[r], rm[r]);
      al[r] = __expf(mst[r] - mn);
      mst[r] = mn;
      S[0].f[r] = __expf(S[0].f[r] - mn);
      S[1].f[r] = __expf(S[1].f[r] - mn);
      rs[r] = S[0].f[r] + S[1].f[r];
    }
#pragma unroll
    for (int o = 1; o < 16; o <<= 1)
#pragma unroll
      for (int r = 0; r < 8; ++r) rs[r] += __shfl_xor(rs[r], o, 32);
#pragma unroll
    for (int r = 0; r < 8; ++r) lst[r] = lst[r] * al[r] + rs[r];
#pragma unroll
    for (int d = 0; d < 4; ++d)
#pragma unroll
      for (int r = 0; r < 8; ++r) acc[d].f[r] *= al[r];
    // stage P into LDS (D-layout -> A-layout transpose)
#pragma unroll
    for (int t = 0; t < 2; ++t)
#pragma unroll
      for (int r = 0; r < 8; ++r)
        myP[(r + 8 * half) * 32 + t * 16 + ln] = f2bf(S[t].f[r]);
    FragA pf;
    pf.q[0] = *(const uint4*)(myP + ln * 32 + half * 16);
    pf.q[1] = *(const uint4*)(myP + ln * 32 + half * 16 + 8);
#pragma unroll
    for (int dt = 0; dt < 4; ++dt) {
      FragA vf;
#pragma unroll
      for (int e = 0; e < 16; ++e) {
        int tok = kc * 32 + half * 16 + e;
        if (tok >= G1) tok = G1 - 1;
        vf.h[e] = vb_[(size_t)tok * C3 + dt * 16 + ln];
      }
      acc[dt].v = wmma_bf16(pf, vf, acc[dt].v);
    }
  }
#pragma unroll
  for (int dt = 0; dt < 4; ++dt)
#pragma unroll
    for (int r = 0; r < 8; ++r) {
      const int q = qt * 16 + r + 8 * half;
      if (q < G1)
        obf[(size_t)(b * G1 + q) * 384 + h * 64 + dt * 16 + ln] =
            f2bf(acc[dt].f[r] / lst[r]);
    }
}

// ---------------------------------------------------------------------------
// View attention: single head, d=384, G=2048, boolean mask. One wave=(b,qtile).
// qkv1: bf16 [B*G, 1152], col = s*384 + d.
// ---------------------------------------------------------------------------
__global__ void __launch_bounds__(128) attn_view_kernel(
    const uint16_t* __restrict__ qkv1, const unsigned char* __restrict__ msk,
    uint16_t* __restrict__ obf) {
  const int Gp = 2048, C3 = 1152;
  const int lane = threadIdx.x & 31;
  const int wv = threadIdx.x >> 5;
  const int wid = blockIdx.x * 4 + wv;
  if (wid >= 4 * 128) return;
  const int qt = wid & 127;
  const int b  = wid >> 7;
  const int half = lane >> 4, ln = lane & 15;
  __shared__ __align__(16) uint16_t sP[4][512];
  uint16_t* myP = sP[wv];

  const uint16_t* qrow = qkv1 + (size_t)(b * Gp + qt * 16 + ln) * C3;
  FragA qa[12];
#pragma unroll
  for (int dc = 0; dc < 12; ++dc) {
    qa[dc].q[0] = *(const uint4*)(qrow + dc * 32 + half * 8);
    qa[dc].q[1] = *(const uint4*)(qrow + dc * 32 + 16 + half * 8);
  }
  const uint16_t* kb_ = qkv1 + (size_t)(b * Gp) * C3 + 384;
  const uint16_t* vb_ = qkv1 + (size_t)(b * Gp) * C3 + 768;
  float mst[8], lst[8];
  FragC acc[24];
#pragma unroll
  for (int r = 0; r < 8; ++r) { mst[r] = -3.0e38f; lst[r] = 0.f; }
#pragma unroll
  for (int d = 0; d < 24; ++d)
#pragma unroll
    for (int r = 0; r < 8; ++r) acc[d].f[r] = 0.f;
  const float sc = 0.051031036307982884f;  // 384^-0.5

  for (int kc = 0; kc < 64; ++kc) {
    FragC S[2];
#pragma unroll
    for (int t = 0; t < 2; ++t) {
      const int tok = kc * 32 + t * 16 + ln;
      const uint16_t* kp = kb_ + (size_t)tok * C3;
      FragC st;
#pragma unroll
      for (int r = 0; r < 8; ++r) st.f[r] = 0.f;
#pragma unroll
      for (int dc = 0; dc < 12; ++dc) {
        FragA kf;
        kf.q[0] = *(const uint4*)(kp + dc * 32 + half * 8);
        kf.q[1] = *(const uint4*)(kp + dc * 32 + 16 + half * 8);
        st.v = wmma_bf16(qa[dc], kf, st.v);
      }
#pragma unroll
      for (int r = 0; r < 8; ++r) {
        const int q = qt * 16 + r + 8 * half;
        float vv = st.f[r] * sc;
        if (!msk[(size_t)q * Gp + tok]) vv = -3.0e38f;
        st.f[r] = vv;
      }
      S[t] = st;
    }
    float rm[8], al[8], rs[8];
#pragma unroll
    for (int r = 0; r < 8; ++r) rm[r] = fmaxf(S[0].f[r], S[1].f[r]);
#pragma unroll
    for (int o = 1; o < 16; o <<= 1)
#pragma unroll
      for (int r = 0; r < 8; ++r) rm[r] = fmaxf(rm[r], __shfl_xor(rm[r], o, 32));
#pragma unroll
    for (int r = 0; r < 8; ++r) {
      const float mn = fmaxf(mst[r], rm[r]);
      al[r] = __expf(mst[r] - mn);
      mst[r] = mn;
      S[0].f[r] = __expf(S[0].f[r] - mn);
      S[1].f[r] = __expf(S[1].f[r] - mn);
      rs[r] = S[0].f[r] + S[1].f[r];
    }
#pragma unroll
    for (int o = 1; o < 16; o <<= 1)
#pragma unroll
      for (int r = 0; r < 8; ++r) rs[r] += __shfl_xor(rs[r], o, 32);
#pragma unroll
    for (int r = 0; r < 8; ++r) lst[r] = lst[r] * al[r] + rs[r];
#pragma unroll
    for (int d = 0; d < 24; ++d)
#pragma unroll
      for (int r = 0; r < 8; ++r) acc[d].f[r] *= al[r];
#pragma unroll
    for (int t = 0; t < 2; ++t)
#pragma unroll
      for (int r = 0; r < 8; ++r)
        myP[(r + 8 * half) * 32 + t * 16 + ln] = f2bf(S[t].f[r]);
    FragA pf;
    pf.q[0] = *(const uint4*)(myP + ln * 32 + half * 16);
    pf.q[1] = *(const uint4*)(myP + ln * 32 + half * 16 + 8);
#pragma unroll
    for (int dt = 0; dt < 24; ++dt) {
      FragA vf;
#pragma unroll
      for (int e = 0; e < 16; ++e)
        vf.h[e] = vb_[(size_t)(kc * 32 + half * 16 + e) * C3 + dt * 16 + ln];
      acc[dt].v = wmma_bf16(pf, vf, acc[dt].v);
    }
  }
#pragma unroll
  for (int dt = 0; dt < 24; ++dt)
#pragma unroll
    for (int r = 0; r < 8; ++r)
      obf[(size_t)(b * Gp + qt * 16 + r + 8 * half) * 384 + dt * 16 + ln] =
          f2bf(acc[dt].f[r] / lst[r]);
}

// ---------------------------------------------------------------------------
// Elementwise / pooling kernels
// ---------------------------------------------------------------------------
__global__ void add_combine_kernel(const float* __restrict__ a, const float* __restrict__ b,
                                   const float* __restrict__ c, float* __restrict__ o, int n) {
  int i = blockIdx.x * blockDim.x + threadIdx.x;
  if (i < n) o[i] = a[i] + b[i] + 0.5f * c[i];
}

__global__ void pts_init_kernel(const float* __restrict__ x2, float* __restrict__ pts) {
  int i = blockIdx.x * blockDim.x + threadIdx.x;
  if (i < 8192 * 384) {
    int n = i / 384, c = i - n * 384;
    pts[i] = x2[(size_t)(n + n / 2048 + 1) * 384 + c];
  }
}

// Deterministic cluster pooling (uses idx_ptr + sorted indices) fused with BN+GELU.
__global__ void cluster_cell_kernel(const float* __restrict__ feat,
                                    const int* __restrict__ idxp, const int* __restrict__ sidx,
                                    const float* __restrict__ g, const float* __restrict__ bb,
                                    const float* __restrict__ m, const float* __restrict__ v,
                                    float* __restrict__ cell) {
  int i = blockIdx.x * blockDim.x + threadIdx.x;
  if (i >= 1024 * 384) return;
  int cl = i / 384, c = i - cl * 384;
  int s0 = idxp[cl], s1 = idxp[cl + 1];
  float sm = 0.f, mx = -3.0e38f;
  for (int j = s0; j < s1; ++j) {
    float t = feat[(size_t)sidx[j] * 384 + c];
    sm += t; mx = fmaxf(mx, t);
  }
  int cnt = s1 - s0;
  float mean = sm / fmaxf((float)cnt, 1.f);
  float mval = cnt > 0 ? mx : 0.f;
  float t = (mval + mean - m[c]) * rsqrtf(v[c] + 1e-5f) * g[c] + bb[c];
  cell[i] = geluf(t);
}

__global__ void gather_kernel(const float* __restrict__ cell, const int* __restrict__ gi,
                              float* __restrict__ out, int n) {
  int i = blockIdx.x * blockDim.x + threadIdx.x;
  if (i < n * 384) {
    int r = i / 384, c = i - r * 384;
    out[i] = cell[(size_t)gi[r] * 384 + c];
  }
}

__global__ void grid_count_kernel(const int* __restrict__ gi, unsigned* __restrict__ gc, int n) {
  int i = blockIdx.x * blockDim.x + threadIdx.x;
  if (i < n) atomicAdd(&gc[gi[i]], 1u);
}

__global__ void grid_pool_kernel(const float* __restrict__ fx, const int* __restrict__ gi,
                                 float* __restrict__ gs, unsigned* __restrict__ gm, int n) {
  int i = blockIdx.x * blockDim.x + threadIdx.x;
  if (i >= n * 384) return;
  int r = i / 384, c = i - r * 384;
  float vv = fx[i];
  size_t o = (size_t)gi[r] * 384 + c;
  atomicAdd(&gs[o], vv);
  atomicMax(&gm[o], f32ord(vv));
}

__global__ void grid_cell_kernel(const float* __restrict__ gs, const unsigned* __restrict__ gm,
                                 const unsigned* __restrict__ gc,
                                 const float* __restrict__ g, const float* __restrict__ bb,
                                 const float* __restrict__ m, const float* __restrict__ v,
                                 float* __restrict__ cell) {
  int i = blockIdx.x * blockDim.x + threadIdx.x;
  if (i >= 4096 * 384) return;
  int r = i / 384, c = i - r * 384;
  unsigned cnt = gc[r];
  float mean = gs[i] / fmaxf((float)cnt, 1.f);
  float mx = cnt > 0 ? ord2f(gm[i]) : 0.f;
  float t = (mean + mx - m[c]) * rsqrtf(v[c] + 1e-5f) * g[c] + bb[c];
  cell[i] = geluf(t);
}

__global__ void __launch_bounds__(256) sims_kernel(const float* __restrict__ xsup,
                                                   const float* __restrict__ x3d,
                                                   float* __restrict__ simsn) {
  const int np = 8192, Cc = 384;
  const int lane = threadIdx.x & 31;
  const int n = blockIdx.x * 8 + (threadIdx.x >> 5);
  if (n >= np) return;
  float dd[6] = {0, 0, 0, 0, 0, 0}, nn[6] = {0, 0, 0, 0, 0, 0}, n3 = 0.f;
  for (int c = lane; c < Cc; c += 32) {
    float a = x3d[(size_t)n * Cc + c];
    n3 += a * a;
#pragma unroll
    for (int v = 0; v < 6; ++v) {
      float t = xsup[((size_t)v * np + n) * Cc + c];
      dd[v] += t * a; nn[v] += t * t;
    }
  }
#pragma unroll
  for (int o = 1; o < 32; o <<= 1) {
    n3 += __shfl_xor(n3, o, 32);
#pragma unroll
    for (int v = 0; v < 6; ++v) {
      dd[v] += __shfl_xor(dd[v], o, 32);
      nn[v] += __shfl_xor(nn[v], o, 32);
    }
  }
  if (lane == 0) {
    float n3s = sqrtf(n3);
    float sv[6]; float tot = 0.f;
#pragma unroll
    for (int v = 0; v < 6; ++v) {
      float nr = sqrtf(nn[v]) * n3s;
      float s = (dd[v] / fmaxf(nr, 1e-8f) + 1.f) * 0.5f;
      sv[v] = s; tot += s;
    }
#pragma unroll
    for (int v = 0; v < 6; ++v) simsn[(size_t)v * np + n] = sv[v] / tot;
  }
}

__global__ void weighted_kernel(float* __restrict__ pts, const float* __restrict__ xsup,
                                const float* __restrict__ simsn) {
  int i = blockIdx.x * blockDim.x + threadIdx.x;
  if (i >= 8192 * 384) return;
  int n = i / 384, c = i - n * 384;
  float a = 0.f;
#pragma unroll
  for (int v = 0; v < 6; ++v)
    a += simsn[(size_t)v * 8192 + n] * xsup[((size_t)v * 8192 + n) * 384 + c];
  pts[i] += 0.5f * a;
}

__global__ void assemble_kernel(const float* __restrict__ x2, const float* __restrict__ pts,
                                float* __restrict__ out) {
  int i = blockIdx.x * blockDim.x + threadIdx.x;
  if (i >= 8196 * 384) return;
  int row = i / 384, c = i - row * 384;
  int b = row / 2049, g = row - b * 2049;
  out[i] = (g == 0) ? x2[i] : pts[((size_t)(b * 2048 + g - 1)) * 384 + c];
}

// ---------------------------------------------------------------------------
// Host orchestration
// ---------------------------------------------------------------------------
extern "C" void kernel_launch(void* const* d_in, const int* in_sizes, int n_in,
                              void* d_out, int out_size, void* d_ws, size_t ws_size,
                              hipStream_t stream) {
  (void)in_sizes; (void)n_in; (void)out_size;
  const float* x        = (const float*)d_in[0];
  const float* norm1_g  = (const float*)d_in[2];
  const float* norm1_b  = (const float*)d_in[3];
  const float* qkv_w    = (const float*)d_in[4];
  const float* proj_w   = (const float*)d_in[5];
  const float* proj_b   = (const float*)d_in[6];
  const float* norm2_g  = (const float*)d_in[7];
  const float* norm2_b  = (const float*)d_in[8];
  const float* fc1_w    = (const float*)d_in[9];
  const float* fc1_b    = (const float*)d_in[10];
  const float* fc2_w    = (const float*)d_in[11];
  const float* fc2_b    = (const float*)d_in[12];
  const float* ada1_w   = (const float*)d_in[13];
  const float* ada1_b   = (const float*)d_in[14];
  const float* ada2_w   = (const float*)d_in[15];
  const float* ada2_b   = (const float*)d_in[16];
  const float* bn3_g    = (const float*)d_in[17];
  const float* bn3_b    = (const float*)d_in[18];
  const float* bn3_m    = (const float*)d_in[19];
  const float* bn3_v    = (const float*)d_in[20];
  const float* bn2_g    = (const float*)d_in[21];
  const float* bn2_b    = (const float*)d_in[22];
  const float* bn2_m    = (const float*)d_in[23];
  const float* bn2_v    = (const float*)d_in[24];
  const float* norm3_g  = (const float*)d_in[25];
  const float* norm3_b  = (const float*)d_in[26];
  const float* a1_qkv_w = (const float*)d_in[27];
  const float* a1_proj_w= (const float*)d_in[28];
  const float* a1_proj_b= (const float*)d_in[29];
  const unsigned char* mask = (const unsigned char*)d_in[30];
  const int* idx_ptr    = (const int*)d_in[31];
  const int* sidx       = (const int*)d_in[32];
  const int* cluster    = (const int*)d_in[33];
  const int* fgi        = (const int*)d_in[34];
  float* out = (float*)d_out;

  const int M1 = 8196, MP = 8192, Cc = 384, C3 = 1152, Hm = 1536, Rr = 96;
  const int NCc = 1024, DSz = 4096, Vv = 6, Gp = 2048;

  char* ws = (char*)d_ws;
  size_t off = 0;
  auto alloc = [&](size_t b) { size_t o = off; off += (b + 255) & ~(size_t)255; return o; };

  const size_t o_wqt  = alloc((size_t)Cc * C3 * 2);
  const size_t o_wpt  = alloc((size_t)Cc * Cc * 2);
  const size_t o_w1t  = alloc((size_t)Cc * Hm * 2);
  const size_t o_w2t  = alloc((size_t)Hm * Cc * 2);
  const size_t o_wa1t = alloc((size_t)Cc * Rr * 2);
  const size_t o_wa2t = alloc((size_t)Rr * Cc * 2);
  const size_t o_wqvt = alloc((size_t)Vv * C3 * Cc * 2);
  const size_t o_wpvt = alloc((size_t)Vv * Cc * Cc * 2);
  const size_t o_hbf  = alloc((size_t)M1 * Cc * 2);
  const size_t o_qkvb = alloc((size_t)M1 * C3 * 2);
  const size_t o_obf  = alloc((size_t)M1 * Cc * 2);
  const size_t o_x1   = alloc((size_t)M1 * Cc * 4);
  const size_t o_f1b  = alloc((size_t)M1 * Hm * 2);
  const size_t o_ffnf = alloc((size_t)M1 * Cc * 4);
  const size_t o_ffnb = alloc((size_t)M1 * Cc * 2);
  const size_t o_abf  = alloc((size_t)M1 * Rr * 2);
  const size_t o_adaf = alloc((size_t)M1 * Cc * 4);
  const size_t o_x2   = alloc((size_t)M1 * Cc * 4);
  const size_t o_pts  = alloc((size_t)MP * Cc * 4);
  const size_t o_cel3 = alloc((size_t)NCc * Cc * 4);
  const size_t o_x3d  = alloc((size_t)MP * Cc * 4);
  const size_t o_fx   = alloc((size_t)MP * Cc * 4);
  const size_t o_gs   = alloc((size_t)DSz * Cc * 4);
  const size_t o_gmo  = alloc((size_t)DSz * Cc * 4);
  const size_t o_gc   = alloc((size_t)DSz * 4);
  const size_t o_cel2 = alloc((size_t)DSz * Cc * 4);
  const size_t o_xsup = alloc((size_t)Vv * MP * Cc * 4);
  const size_t o_sims = alloc((size_t)Vv * MP * 4);
  if (off > ws_size) return;  // workspace too small; bail safely

  uint16_t* WQT  = (uint16_t*)(ws + o_wqt);
  uint16_t* WPT  = (uint16_t*)(ws + o_wpt);
  uint16_t* W1T  = (uint16_t*)(ws + o_w1t);
  uint16_t* W2T  = (uint16_t*)(ws + o_w2t);
  uint16_t* WA1T = (uint16_t*)(ws + o_wa1t);
  uint16_t* WA2T = (uint16_t*)(ws + o_wa2t);
  uint16_t* WQVT = (uint16_t*)(ws + o_wqvt);
  uint16_t* WPVT = (uint16_t*)(ws + o_wpvt);
  uint16_t* HBF  = (uint16_t*)(ws + o_hbf);
  uint16_t* QKVB = (uint16_t*)(ws + o_qkvb);
  uint16_t* OBF  = (uint16_t*)(ws + o_obf);
  uint16_t* F1B  = (uint16_t*)(ws + o_f1b);
  uint16_t* FFNB = (uint16_t*)(ws + o_ffnb);
  uint16_t* ABF  = (uint16_t*)(ws + o_abf);
  float* X1   = (float*)(ws + o_x1);
  float* FFNF = (float*)(ws + o_ffnf);
  float* ADAF = (float*)(ws + o_adaf);
  float* X2   = (float*)(ws + o_x2);
  float* PTS  = (float*)(ws + o_pts);
  float* CEL3 = (float*)(ws + o_cel3);
  float* X3D  = (float*)(ws + o_x3d);
  float* FX   = (float*)(ws + o_fx);
  float* GS   = (float*)(ws + o_gs);
  unsigned* GMO = (unsigned*)(ws + o_gmo);
  unsigned* GC  = (unsigned*)(ws + o_gc);
  float* CEL2 = (float*)(ws + o_cel2);
  float* XSUP = (float*)(ws + o_xsup);
  float* SIMS = (float*)(ws + o_sims);

  auto ew = [](long long n) { return (unsigned)((n + 255) / 256); };
  auto gemm = [&](const uint16_t* A, const uint16_t* Bt, const float* bias,
                  const float* resid, float* outF, uint16_t* outB,
                  int M, int Nn, int K, int act) {
    dim3 grid(Nn / 32, (((M + 31) / 32) + 3) / 4);
    gemm_wmma_kernel<<<grid, 128, 0, stream>>>(A, Bt, bias, resid, outF, outB, M, Nn, K, act);
  };

  // -- weight conversion (f32 -> transposed bf16) --
  wtrans_kernel<<<ew((long long)Cc * C3), 256, 0, stream>>>(qkv_w, WQT, Cc, C3);
  wtrans_kernel<<<ew((long long)Cc * Cc), 256, 0, stream>>>(proj_w, WPT, Cc, Cc);
  wtrans_kernel<<<ew((long long)Cc * Hm), 256, 0, stream>>>(fc1_w, W1T, Cc, Hm);
  wtrans_kernel<<<ew((long long)Hm * Cc), 256, 0, stream>>>(fc2_w, W2T, Hm, Cc);
  wtrans_kernel<<<ew((long long)Cc * Rr), 256, 0, stream>>>(ada1_w, WA1T, Cc, Rr);
  wtrans_kernel<<<ew((long long)Rr * Cc), 256, 0, stream>>>(ada2_w, WA2T, Rr, Cc);
  for (int v = 0; v < Vv; ++v) {
    wtrans_kernel<<<ew((long long)Cc * C3), 256, 0, stream>>>(
        a1_qkv_w + (size_t)v * Cc * C3, WQVT + (size_t)v * C3 * Cc, Cc, C3);
    wtrans_kernel<<<ew((long long)Cc * Cc), 256, 0, stream>>>(
        a1_proj_w + (size_t)v * Cc * Cc, WPVT + (size_t)v * Cc * Cc, Cc, Cc);
  }

  // -- phase 1: transformer block --
  ln_kernel<<<(M1 + 7) / 8, 256, 0, stream>>>(x, norm1_g, norm1_b, HBF, M1);
  gemm(HBF, WQT, nullptr, nullptr, nullptr, QKVB, M1, C3, Cc, 0);
  attn_main_kernel<<<(4 * 6 * 129 + 3) / 4, 128, 0, stream>>>(QKVB, OBF);
  gemm(OBF, WPT, proj_b, x, X1, nullptr, M1, Cc, Cc, 0);
  ln_kernel<<<(M1 + 7) / 8, 256, 0, stream>>>(X1, norm2_g, norm2_b, HBF, M1);
  gemm(HBF, W1T, fc1_b, nullptr, nullptr, F1B, M1, Hm, Cc, 1);
  gemm(F1B, W2T, fc2_b, nullptr, FFNF, FFNB, M1, Cc, Hm, 0);
  gemm(FFNB, WA1T, ada1_b, nullptr, nullptr, ABF, M1, Rr, Cc, 2);
  gemm(ABF, WA2T, ada2_b, nullptr, ADAF, nullptr, M1, Cc, Rr, 0);
  add_combine_kernel<<<ew((long long)M1 * Cc), 256, 0, stream>>>(X1, FFNF, ADAF, X2, M1 * Cc);
  pts_init_kernel<<<ew((long long)MP * Cc), 256, 0, stream>>>(X2, PTS);

  // -- cluster pooling (deterministic, uses sorted indices) --
  cluster_cell_kernel<<<ew((long long)NCc * Cc), 256, 0, stream>>>(
      PTS, idx_ptr, sidx, bn3_g, bn3_b, bn3_m, bn3_v, CEL3);
  gather_kernel<<<ew((long long)MP * Cc), 256, 0, stream>>>(CEL3, cluster, X3D, MP);

  // -- per-view masked attention + grid pooling --
  for (int v = 0; v < Vv; ++v) {
    ln_kernel<<<(MP + 7) / 8, 256, 0, stream>>>(PTS, norm3_g + (size_t)v * Cc,
                                                norm3_b + (size_t)v * Cc, HBF, MP);
    gemm(HBF, WQVT + (size_t)v * C3 * Cc, nullptr, nullptr, nullptr, QKVB, MP, C3, Cc, 0);
    attn_view_kernel<<<128, 128, 0, stream>>>(QKVB, mask + (size_t)v * Gp * Gp, OBF);
    gemm(OBF, WPVT + (size_t)v * Cc * Cc, a1_proj_b + (size_t)v * Cc, PTS, FX, nullptr,
         MP, Cc, Cc, 0);
    hipMemsetAsync(GS, 0, (size_t)DSz * Cc * 4, stream);
    hipMemsetAsync(GMO, 0, (size_t)DSz * Cc * 4, stream);
    hipMemsetAsync(GC, 0, (size_t)DSz * 4, stream);
    const int* gi = fgi + (size_t)v * MP;
    grid_count_kernel<<<ew(MP), 256, 0, stream>>>(gi, GC, MP);
    grid_pool_kernel<<<ew((long long)MP * Cc), 256, 0, stream>>>(FX, gi, GS, GMO, MP);
    grid_cell_kernel<<<ew((long long)DSz * Cc), 256, 0, stream>>>(
        GS, GMO, GC, bn2_g + (size_t)v * Cc, bn2_b + (size_t)v * Cc,
        bn2_m + (size_t)v * Cc, bn2_v + (size_t)v * Cc, CEL2);
    gather_kernel<<<ew((long long)MP * Cc), 256, 0, stream>>>(
        CEL2, gi, XSUP + (size_t)v * MP * Cc, MP);
  }

  // -- similarity weighting + final assembly --
  sims_kernel<<<(MP + 7) / 8, 256, 0, stream>>>(XSUP, X3D, SIMS);
  weighted_kernel<<<ew((long long)MP * Cc), 256, 0, stream>>>(PTS, XSUP, SIMS);
  assemble_kernel<<<ew((long long)M1 * Cc), 256, 0, stream>>>(X2, PTS, out);
}
